// CausalSelfAttention_51505247813856
// MI455X (gfx1250) — matrix operand order repaired and checked by reading, not compiled, not run on hardware
//
#include <hip/hip_runtime.h>
#include <stdint.h>

// ---------------- CDNA5 (gfx1250) causal self-attention, bf16 WMMA ----------
// B=4, T=2048, C=1024, H=16, D=64.
// Pipeline: f32->bf16 convert | QKV GEMM (WMMA, double-buffered async-LDS
// staging, Q pre-scaled by 1/sqrt(D)) -> per-head Q/K/V | flash attention
// (double-buffered K/V staging, WMMA QK^T + DPP online softmax + WMMA PV,
// per-wave causal tile skipping) | proj GEMM -> f32.

#define B_ 4
#define T_ 2048
#define C_ 1024
#define NH 16
#define HD 64

typedef __attribute__((ext_vector_type(16))) __bf16 v16bf;
typedef __attribute__((ext_vector_type(8)))  float  v8f;

union FragB { v16bf v; uint4 u[2]; };

__device__ __forceinline__ unsigned short f32_to_bf16(float f) {
  union { float f; uint32_t u; } x; x.f = f;
  uint32_t u = x.u;
  uint32_t r = u + 0x7FFFu + ((u >> 16) & 1u);      // round-to-nearest-even
  if ((u & 0x7F800000u) == 0x7F800000u) r = u;      // inf/nan: truncate
  return (unsigned short)(r >> 16);
}

// 16-lane (DPP row) commutative reductions: xor1, xor2, half_mirror, mirror.
__device__ __forceinline__ float row_max16(float x) {
  union { float f; int i; } a, b;
  a.f = x;
  b.i = __builtin_amdgcn_update_dpp(a.i, a.i, 0xB1,  0xF, 0xF, true); a.f = fmaxf(a.f, b.f);
  b.i = __builtin_amdgcn_update_dpp(a.i, a.i, 0x4E,  0xF, 0xF, true); a.f = fmaxf(a.f, b.f);
  b.i = __builtin_amdgcn_update_dpp(a.i, a.i, 0x141, 0xF, 0xF, true); a.f = fmaxf(a.f, b.f);
  b.i = __builtin_amdgcn_update_dpp(a.i, a.i, 0x140, 0xF, 0xF, true); a.f = fmaxf(a.f, b.f);
  return a.f;
}
__device__ __forceinline__ float row_sum16(float x) {
  union { float f; int i; } a, b;
  a.f = x;
  b.i = __builtin_amdgcn_update_dpp(a.i, a.i, 0xB1,  0xF, 0xF, true); a.f += b.f;
  b.i = __builtin_amdgcn_update_dpp(a.i, a.i, 0x4E,  0xF, 0xF, true); a.f += b.f;
  b.i = __builtin_amdgcn_update_dpp(a.i, a.i, 0x141, 0xF, 0xF, true); a.f += b.f;
  b.i = __builtin_amdgcn_update_dpp(a.i, a.i, 0x140, 0xF, 0xF, true); a.f += b.f;
  return a.f;
}

// Async global->LDS 32B/thread copy (two b128; the instruction offset applies
// to both the LDS and the memory address per ISA pseudocode). ASYNCcnt.
__device__ __forceinline__ void async_copy_32B(unsigned lds_addr,
                                               unsigned long long gaddr) {
  asm volatile("global_load_async_to_lds_b128 %0, %1, off\n\t"
               "global_load_async_to_lds_b128 %0, %1, off offset:16"
               :: "v"(lds_addr), "v"(gaddr) : "memory");
}
__device__ __forceinline__ void wait_async_le4() {
  asm volatile("s_wait_asynccnt 0x4" ::: "memory");
}
__device__ __forceinline__ void wait_async0() {
  asm volatile("s_wait_asynccnt 0x0" ::: "memory");
}

// ------------------------------- convert ------------------------------------
__global__ void k_f32_to_bf16(const float* __restrict__ in,
                              unsigned short* __restrict__ out, int n) {
  int i = blockIdx.x * blockDim.x + threadIdx.x;
  int stride = gridDim.x * blockDim.x;
  for (; i < n; i += stride) out[i] = f32_to_bf16(in[i]);
}

// ------------------------------- GEMM ---------------------------------------
// out[m,n] = sum_k A[m,k] * W[n,k]  (A:[M][1024], W:[N][1024], bf16 row-major)
// 256 thr (8 waves), tile 128x128, K-step 32, double-buffered async staging.
// MODE 0: scatter bf16 to q/k/v [B,H,T,D], Q scaled by 0.125.
// MODE 1: f32 out [M][1024].
template <int MODE>
__global__ __launch_bounds__(256, 1) void k_gemm(
    const unsigned short* __restrict__ A,
    const unsigned short* __restrict__ W,
    unsigned short* __restrict__ q_out,
    unsigned short* __restrict__ k_out,
    unsigned short* __restrict__ v_out,
    float* __restrict__ f_out) {
  __shared__ __align__(16) unsigned short As[2][128][40];  // 32 data + 8 pad
  __shared__ __align__(16) unsigned short Bs[2][128][40];
  const int lane = threadIdx.x & 31;
  const int wave = threadIdx.x >> 5;
  const int wm = wave >> 2, wn = wave & 3;
  const int half = lane >> 4, l15 = lane & 15;
  const int m0 = blockIdx.y * 128, n0 = blockIdx.x * 128;

  v8f acc[4][2] = {};
  const int r  = threadIdx.x >> 1;   // staged row 0..127
  const int hh = threadIdx.x & 1;    // 16-element half of the 32-wide row

  const unsigned ldsA[2] = {(unsigned)(uintptr_t)&As[0][r][16 * hh],
                            (unsigned)(uintptr_t)&As[1][r][16 * hh]};
  const unsigned ldsB[2] = {(unsigned)(uintptr_t)&Bs[0][r][16 * hh],
                            (unsigned)(uintptr_t)&Bs[1][r][16 * hh]};
  const unsigned short* gA = A + (size_t)(m0 + r) * C_ + 16 * hh;
  const unsigned short* gB = W + (size_t)(n0 + r) * C_ + 16 * hh;

  auto stage = [&](int buf, int k0) {
    async_copy_32B(ldsA[buf], (unsigned long long)(uintptr_t)(gA + k0));
    async_copy_32B(ldsB[buf], (unsigned long long)(uintptr_t)(gB + k0));
  };

  const int NSTEP = C_ / 32;
  stage(0, 0);
  for (int s = 0; s < NSTEP; ++s) {
    const int cur = s & 1;
    if (s + 1 < NSTEP) {
      stage(cur ^ 1, (s + 1) * 32);      // prefetch next tile (other buffer)
      wait_async_le4();                  // oldest tile (4 async ops) complete
    } else {
      wait_async0();
    }
    __syncthreads();                     // all waves' copies for tile s landed

    FragB afr[4], bfr[2];
#pragma unroll
    for (int i = 0; i < 4; ++i) {                      // A layout: row=l15
      int row = wm * 64 + i * 16 + l15;                // K chunks: 8h, 16+8h
      afr[i].u[0] = *(const uint4*)&As[cur][row][8 * half];
      afr[i].u[1] = *(const uint4*)&As[cur][row][16 + 8 * half];
    }
#pragma unroll
    for (int j = 0; j < 2; ++j) {                      // B layout: col=l15
      int row = wn * 32 + j * 16 + l15;                // K chunks: 8h, 16+8h
      bfr[j].u[0] = *(const uint4*)&Bs[cur][row][8 * half];
      bfr[j].u[1] = *(const uint4*)&Bs[cur][row][16 + 8 * half];
    }
#pragma unroll
    for (int i = 0; i < 4; ++i)
#pragma unroll
      for (int j = 0; j < 2; ++j)
        acc[i][j] = __builtin_amdgcn_wmma_f32_16x16x32_bf16(
            false, afr[i].v, false, bfr[j].v, (short)0, acc[i][j], false, false);
    __syncthreads();                     // tile s consumed; buffer reusable
  }

#pragma unroll
  for (int i = 0; i < 4; ++i)
#pragma unroll
    for (int j = 0; j < 2; ++j) {
      int n = n0 + wn * 32 + j * 16 + l15;
      if (MODE == 0) {
        int which = n >> 10, hd = n & 1023, h = hd >> 6, d = hd & 63;
        unsigned short* dst = which == 0 ? q_out : (which == 1 ? k_out : v_out);
        float sc = (which == 0) ? 0.125f : 1.0f;   // fold 1/sqrt(64) into Q
#pragma unroll
        for (int v = 0; v < 8; ++v) {
          int m = m0 + wm * 64 + i * 16 + v + 8 * half;
          int b = m >> 11, t = m & (T_ - 1);
          dst[(size_t)((b * NH + h) * T_ + t) * HD + d] =
              f32_to_bf16(acc[i][j][v] * sc);
        }
      } else {
#pragma unroll
        for (int v = 0; v < 8; ++v) {
          int m = m0 + wm * 64 + i * 16 + v + 8 * half;
          f_out[(size_t)m * C_ + n] = acc[i][j][v];
        }
      }
    }
}

// ---------------------------- flash attention -------------------------------
// Grid: (T/128, B*H). Block 256 thr, 8 waves x 16 query rows.
// Double-buffered 64-key tiles: async K -> LDS, V -> regs -> transposed LDS,
// next tile's staging overlapped with current tile's WMMAs. Waves skip
// compute for tiles fully above their causal boundary (barriers stay uniform).
__global__ __launch_bounds__(256, 1) void k_flash(
    const unsigned short* __restrict__ Q,   // pre-scaled by 1/sqrt(D)
    const unsigned short* __restrict__ K,
    const unsigned short* __restrict__ V,
    unsigned short* __restrict__ O) {       // [B*T][C] bf16
  __shared__ __align__(16) unsigned short Ks[2][64][72];   // 64 data + 8 pad
  __shared__ __align__(16) unsigned short Vt[2][64][72];   // transposed V
  __shared__ __align__(16) unsigned short Ps[8][16][72];   // per-wave P scratch
  const int lane = threadIdx.x & 31, wave = threadIdx.x >> 5;
  const int half = lane >> 4, l15 = lane & 15;
  const int qblk = blockIdx.x;
  const int bh = blockIdx.y;
  const int b = bh >> 4, h = bh & 15;
  const size_t base = (size_t)bh * T_ * HD;
  const int qbase = qblk * 128 + wave * 16;
  const int qmax = qbase + 15;

  FragB qf[2];
  {
    const unsigned short* qrow = Q + base + (size_t)(qbase + l15) * HD;
#pragma unroll
    for (int s = 0; s < 2; ++s) {
      qf[s].u[0] = *(const uint4*)(qrow + 32 * s + 8 * half);
      qf[s].u[1] = *(const uint4*)(qrow + 32 * s + 16 + 8 * half);
    }
  }

  float mrow[8], lrow[8];
#pragma unroll
  for (int v = 0; v < 8; ++v) { mrow[v] = -INFINITY; lrow[v] = 0.f; }
  v8f acco[4] = {};
  const int nkt = (qblk + 1) * 2;    // 64-key tiles up to (qblk+1)*128 keys

  const int rr = threadIdx.x >> 2;          // staged key row 0..63
  const int cc = (threadIdx.x & 3) * 16;    // dim chunk 0,16,32,48
  const unsigned ldsK[2] = {(unsigned)(uintptr_t)&Ks[0][rr][cc],
                            (unsigned)(uintptr_t)&Ks[1][rr][cc]};
  const unsigned short* gK = K + base + (size_t)rr * HD + cc;
  const unsigned short* gV = V + base + (size_t)rr * HD + cc;

  auto storeVt = [&](int buf, uint4 a, uint4 c) {
    const unsigned short* vs = (const unsigned short*)&a;
#pragma unroll
    for (int e = 0; e < 8; ++e) Vt[buf][cc + e][rr] = vs[e];
    vs = (const unsigned short*)&c;
#pragma unroll
    for (int e = 0; e < 8; ++e) Vt[buf][cc + 8 + e][rr] = vs[e];
  };

  // prologue: stage tile 0 into buffer 0
  async_copy_32B(ldsK[0], (unsigned long long)(uintptr_t)gK);
  storeVt(0, *(const uint4*)gV, *(const uint4*)(gV + 8));
  wait_async0();
  __syncthreads();

  for (int kt = 0; kt < nkt; ++kt) {
    const int buf = kt & 1;
    const int kb = kt * 64;
    const bool more = (kt + 1 < nkt);
    uint4 nv0, nv1;
    if (more) {                          // overlap next tile's staging
      async_copy_32B(ldsK[buf ^ 1],
          (unsigned long long)(uintptr_t)(gK + (size_t)(kb + 64) * HD));
      nv0 = *(const uint4*)(gV + (size_t)(kb + 64) * HD);
      nv1 = *(const uint4*)(gV + (size_t)(kb + 64) * HD + 8);
    }

    if (kb <= qmax) {                    // wave-uniform causal skip
      // S = Q K^T : 4 key subtiles x 2 d-steps
      v8f accs[4] = {};
#pragma unroll
      for (int j = 0; j < 4; ++j) {
        int krow = j * 16 + l15;
#pragma unroll
        for (int s = 0; s < 2; ++s) {
          FragB kf;
          kf.u[0] = *(const uint4*)&Ks[buf][krow][32 * s + 8 * half];
          kf.u[1] = *(const uint4*)&Ks[buf][krow][32 * s + 16 + 8 * half];
          accs[j] = __builtin_amdgcn_wmma_f32_16x16x32_bf16(
              false, qf[s].v, false, kf.v, (short)0, accs[j], false, false);
        }
      }

      // DPP online softmax; specialize masked vs interior tiles
      float fac[8];
      auto softmax_step = [&](bool masked) {
#pragma unroll
        for (int v = 0; v < 8; ++v) {
          int q = qbase + v + 8 * half;
          int rowi = v + 8 * half;
          float sv[4];
#pragma unroll
          for (int j = 0; j < 4; ++j) {
            sv[j] = accs[j][v];                   // scale pre-folded into Q
            if (masked && (kb + j * 16 + l15 > q)) sv[j] = -INFINITY;
          }
          float mx = fmaxf(fmaxf(sv[0], sv[1]), fmaxf(sv[2], sv[3]));
          mx = row_max16(mx);
          float mnew = fmaxf(mrow[v], mx);
          float rs = 0.f;
#pragma unroll
          for (int j = 0; j < 4; ++j) {
            float p = __expf(sv[j] - mnew);
            rs += p;
            Ps[wave][rowi][j * 16 + l15] = f32_to_bf16(p);
          }
          rs = row_sum16(rs);
          float f = __expf(mrow[v] - mnew);
          lrow[v] = lrow[v] * f + rs;
          mrow[v] = mnew;
          fac[v] = f;
        }
      };
      if (kb + 63 <= qbase) softmax_step(false);   // interior: no masking
      else                  softmax_step(true);    // diagonal tile
      // intra-wave LDS store->load ordering for the P layout shuffle
      asm volatile("s_wait_dscnt 0x0" ::: "memory");

#pragma unroll
      for (int j = 0; j < 4; ++j)
#pragma unroll
        for (int v = 0; v < 8; ++v) acco[j][v] *= fac[v];

      // O += P @ V : 2 key-steps x 4 dim subtiles
#pragma unroll
      for (int s = 0; s < 2; ++s) {
        FragB pf;
        pf.u[0] = *(const uint4*)&Ps[wave][l15][32 * s + 8 * half];
        pf.u[1] = *(const uint4*)&Ps[wave][l15][32 * s + 16 + 8 * half];
#pragma unroll
        for (int j = 0; j < 4; ++j) {
          FragB vf;
          int vrow = j * 16 + l15;
          vf.u[0] = *(const uint4*)&Vt[buf][vrow][32 * s + 8 * half];
          vf.u[1] = *(const uint4*)&Vt[buf][vrow][32 * s + 16 + 8 * half];
          acco[j] = __builtin_amdgcn_wmma_f32_16x16x32_bf16(
              false, pf.v, false, vf.v, (short)0, acco[j], false, false);
        }
      }
    }

    if (more) {
      storeVt(buf ^ 1, nv0, nv1);        // other buffer: no read conflict
      wait_async0();                     // next K tile landed
    }
    __syncthreads();                     // uniform across all waves
  }

  float inv[8];
#pragma unroll
  for (int v = 0; v < 8; ++v) inv[v] = __builtin_amdgcn_rcpf(lrow[v]);
  const size_t orow = (size_t)b * T_ + qbase;
#pragma unroll
  for (int j = 0; j < 4; ++j) {
    int col = h * HD + j * 16 + l15;
#pragma unroll
    for (int v = 0; v < 8; ++v) {
      int mr = v + 8 * half;
      O[(orow + mr) * C_ + col] = f32_to_bf16(acco[j][v] * inv[v]);
    }
  }
}

// ------------------------------- launch -------------------------------------
extern "C" void kernel_launch(void* const* d_in, const int* in_sizes, int n_in,
                              void* d_out, int out_size, void* d_ws, size_t ws_size,
                              hipStream_t stream) {
  (void)in_sizes; (void)n_in; (void)out_size; (void)ws_size;
  const float* x      = (const float*)d_in[0];
  const float* w_qkv  = (const float*)d_in[1];
  const float* w_proj = (const float*)d_in[2];
  float* out = (float*)d_out;

  char* ws = (char*)d_ws;
  size_t off = 0;
  auto alloc = [&](size_t bytes) {
    char* p = ws + off;
    off += (bytes + 255) & ~(size_t)255;
    return p;
  };
  unsigned short* xb     = (unsigned short*)alloc((size_t)B_ * T_ * C_ * 2);
  unsigned short* wqkvb  = (unsigned short*)alloc((size_t)3 * C_ * C_ * 2);
  unsigned short* wprojb = (unsigned short*)alloc((size_t)C_ * C_ * 2);
  unsigned short* qb     = (unsigned short*)alloc((size_t)B_ * T_ * C_ * 2);
  unsigned short* kb     = (unsigned short*)alloc((size_t)B_ * T_ * C_ * 2);
  unsigned short* vb     = (unsigned short*)alloc((size_t)B_ * T_ * C_ * 2);
  unsigned short* attout = (unsigned short*)alloc((size_t)B_ * T_ * C_ * 2);

  k_f32_to_bf16<<<dim3(2048), dim3(256), 0, stream>>>(x, xb, B_ * T_ * C_);
  k_f32_to_bf16<<<dim3(1024), dim3(256), 0, stream>>>(w_qkv, wqkvb, 3 * C_ * C_);
  k_f32_to_bf16<<<dim3(512),  dim3(256), 0, stream>>>(w_proj, wprojb, C_ * C_);

  // QKV projection: M=8192, N=3072
  k_gemm<0><<<dim3(3 * C_ / 128, B_ * T_ / 128), dim3(256), 0, stream>>>(
      xb, wqkvb, qb, kb, vb, nullptr);

  // Flash attention: (T/128, B*H)
  k_flash<<<dim3(T_ / 128, B_ * NH), dim3(256), 0, stream>>>(qb, kb, vb, attout);

  // Output projection: M=8192, N=1024, f32 out
  k_gemm<1><<<dim3(C_ / 128, B_ * T_ / 128), dim3(256), 0, stream>>>(
      attout, wprojb, nullptr, nullptr, nullptr, out);
}
